// MyDMPNNEncoderLayer_54030688584201
// MI455X (gfx1250) — matrix operand order repaired and checked by reading, compiled, not run on hardware
//
#include <hip/hip_runtime.h>
#include <stdint.h>

#define NUM_ATOMS   65536
#define NUM_BONDS   131072
#define NB1         (NUM_BONDS + 1)
#define ATOM_FDIM   133
#define CONCAT_FDIM 147
#define D_HIDDEN    300
#define GF_DIM      2048
#define N_MOLS      4096
#define OUT_DIM     (D_HIDDEN + GF_DIM)   // 2348

#define H_STRIDE    304     // bf16 row stride for inp/h (16B-aligned rows, cols 300..303 kept zero)
#define NT_TILES    19      // 300 -> 19 tiles of 16
#define BROWS       96      // bond rows per block (6 m-tiles)

// GEMM K configs: K padded to multiple of 32; strides padded for LDS bank-conflict freedom (+16B alignment)
#define K1_STRIDE 168       // K=160 (147 padded), 5 ksteps
#define K2_STRIDE 328       // K=320 (300 padded), 10 ksteps
#define K3_STRIDE 456       // K=448 (433 padded), 14 ksteps

typedef __attribute__((ext_vector_type(16))) __bf16 v16bf;
typedef __attribute__((ext_vector_type(8)))  float  v8f;

union FragU { uint4 q[2]; v16bf v; };

__device__ __forceinline__ float bf2f(unsigned short u) {
    unsigned int x = ((unsigned int)u) << 16;
    return __builtin_bit_cast(float, x);
}
__device__ __forceinline__ unsigned short f2bf(float f) {
    unsigned int x = __builtin_bit_cast(unsigned int, f);
    x = x + 0x7fffu + ((x >> 16) & 1u);   // round-to-nearest-even
    return (unsigned short)(x >> 16);
}
// packed 2x f32 -> 2x bf16 in one dword
#if defined(__has_builtin)
#if __has_builtin(__builtin_amdgcn_cvt_pk_bf16_f32)
#define HAVE_PK_BF16 1
#endif
#endif
__device__ __forceinline__ unsigned int f2bf_pk(float a, float b) {
#ifdef HAVE_PK_BF16
    auto r = __builtin_amdgcn_cvt_pk_bf16_f32(a, b);
    unsigned int u; __builtin_memcpy(&u, &r, 4); return u;
#else
    return (unsigned int)f2bf(a) | ((unsigned int)f2bf(b) << 16);
#endif
}
// unpack one dword (2 bf16) and accumulate into two floats
__device__ __forceinline__ void bf2_acc(unsigned int w, float& lo, float& hi) {
    lo += __builtin_bit_cast(float, w << 16);
    hi += __builtin_bit_cast(float, w & 0xffff0000u);
}

// A fragment (16-bit A, 16x32): lanes 0-15 hold M=0..15 / K-low, lanes 16-31 M=0..15 / K-high.
__device__ __forceinline__ v16bf load_a_frag(const unsigned short* A, int stride, int mbase,
                                             int kbase, int lane) {
    int row = mbase + (lane & 15);
    int hi  = lane >> 4;
    const unsigned short* p = A + row * stride + kbase + hi * 8;
    FragU u;
    u.q[0] = *(const uint4*)(p);        // K = kbase + hi*8 .. +7
    u.q[1] = *(const uint4*)(p + 16);   // K = kbase + 16 + hi*8 .. +7
    return u.v;
}
// B fragment (16-bit B, 32x16): lane holds column n; lanes0-15 K=kbase+0..7,16..23, lanes16-31 K=kbase+8..15,24..31.
__device__ __forceinline__ v16bf load_b_frag(const unsigned short* W, int kstride, int ntile,
                                             int kbase, int lane) {
    int n  = ntile * 16 + (lane & 15);
    int hi = lane >> 4;
    const unsigned short* p = W + (size_t)n * kstride + kbase + hi * 8;
    FragU u;
    u.q[0] = *(const uint4*)(p);
    u.q[1] = *(const uint4*)(p + 16);
    return u.v;
}

// ---------- weight prep: f32 -> bf16, padded, B-layout ----------
__global__ __launch_bounds__(256) void prep_weights(const float* Wi, const float* Wh, const float* Wo,
                                                    unsigned short* WbI, unsigned short* WbH,
                                                    unsigned short* WbO) {
    int idx = blockIdx.x * 256 + threadIdx.x;
    if (idx < 304 * K1_STRIDE) {
        int n = idx / K1_STRIDE, k = idx % K1_STRIDE;
        WbI[idx] = (n < D_HIDDEN && k < CONCAT_FDIM) ? f2bf(Wi[(size_t)n * CONCAT_FDIM + k]) : (unsigned short)0;
    }
    if (idx < 304 * K2_STRIDE) {
        int n = idx / K2_STRIDE, k = idx % K2_STRIDE;
        WbH[idx] = (n < D_HIDDEN && k < D_HIDDEN) ? f2bf(Wh[(size_t)n * D_HIDDEN + k]) : (unsigned short)0;
    }
    if (idx < 304 * K3_STRIDE) {
        int n = idx / K3_STRIDE, k = idx % K3_STRIDE;
        // K permuted: k<300 -> messages (W_o col 133+k) ; 300<=k<433 -> atom feats (W_o col k-300)
        float v = 0.f; bool ok = false;
        if (n < D_HIDDEN) {
            if (k < D_HIDDEN) { v = Wo[(size_t)n * 433 + 133 + k];   ok = true; }
            else if (k < 433) { v = Wo[(size_t)n * 433 + (k - 300)]; ok = true; }
        }
        WbO[idx] = ok ? f2bf(v) : (unsigned short)0;
    }
}

// ---------- kernel 1: inp = f_ini @ W_i^T ; h = relu(inp) ----------
__global__ __launch_bounds__(256) void k1_init(const float* f_ini, const unsigned short* WbI,
                                               unsigned short* inp, unsigned short* h) {
    __shared__ __align__(16) unsigned short Alds[BROWS * K1_STRIDE];
    const int tid = threadIdx.x, blk = blockIdx.x;
    const bool full = (blk * BROWS + BROWS) <= NB1;   // uniform: all rows valid except last block

    for (int task = tid; task < BROWS * 20; task += 256) {  // 20 chunks of 8 cover K=0..159
        int r = task / 20, c = task % 20;
        int g = blk * BROWS + r;
        bool rowok = full || (g < NB1);
        const float* src = f_ini + (size_t)g * CONCAT_FDIM + c * 8;
        alignas(16) unsigned int o[4];
        if (rowok && c < 18) {          // cols 0..143: unguarded
            #pragma unroll
            for (int e = 0; e < 4; e++) o[e] = f2bf_pk(src[2 * e], src[2 * e + 1]);
        } else {
            float t[8];
            #pragma unroll
            for (int k = 0; k < 8; k++) {
                int col = c * 8 + k;
                t[k] = (rowok && col < CONCAT_FDIM) ? src[k] : 0.f;
            }
            #pragma unroll
            for (int e = 0; e < 4; e++) o[e] = f2bf_pk(t[2 * e], t[2 * e + 1]);
        }
        *(uint4*)&Alds[r * K1_STRIDE + c * 8] = *(const uint4*)o;
    }
    __syncthreads();

    const int wave = tid >> 5, lane = tid & 31;
    for (int nt = wave; nt < NT_TILES; nt += 8) {
        v16bf b[5];
        #pragma unroll
        for (int k = 0; k < 5; k++) b[k] = load_b_frag(WbI, K1_STRIDE, nt, k * 32, lane);
        for (int mt = 0; mt < BROWS / 16; mt++) {
            v8f acc = {};
            #pragma unroll
            for (int k = 0; k < 5; k++) {
                v16bf a = load_a_frag(Alds, K1_STRIDE, mt * 16, k * 32, lane);
                acc = __builtin_amdgcn_wmma_f32_16x16x32_bf16(false, a, false, b[k],
                                                              (short)0, acc, false, false);
            }
            int col = nt * 16 + (lane & 15);
            int hi  = lane >> 4;
            int row0 = blk * BROWS + mt * 16 + hi * 8;     // lanes0-15: M=rr ; lanes16-31: M=rr+8
            size_t base = (size_t)row0 * H_STRIDE + col;
            if (full) {
                #pragma unroll
                for (int rr = 0; rr < 8; rr++) {
                    size_t off = base + (size_t)rr * H_STRIDE;
                    float v = acc[rr];                      // cols>=300 naturally 0 (B cols zeroed)
                    inp[off] = f2bf(v);
                    h[off]   = f2bf(fmaxf(v, 0.f));
                }
            } else {
                #pragma unroll
                for (int rr = 0; rr < 8; rr++) {
                    if (row0 + rr < NB1) {
                        size_t off = base + (size_t)rr * H_STRIDE;
                        float v = acc[rr];
                        inp[off] = f2bf(v);
                        h[off]   = f2bf(fmaxf(v, 0.f));
                    }
                }
            }
        }
    }
}

// ---------- kernel 2: h_out = relu(inp + (gather-sum h_in) @ W_h^T) ----------
__global__ __launch_bounds__(256) void k2_msg(const int* mapping, const unsigned short* h_in,
                                              const unsigned short* inp, const unsigned short* WbH,
                                              unsigned short* h_out) {
    __shared__ __align__(16) unsigned short Alds[BROWS * K2_STRIDE];
    __shared__ int4 Map[BROWS];
    const int tid = threadIdx.x, blk = blockIdx.x;
    const bool full = (blk * BROWS + BROWS) <= NB1;

    if (tid < BROWS) {
        int g = blk * BROWS + tid;
        Map[tid] = (full || g < NB1) ? ((const int4*)mapping)[g] : make_int4(-1, -1, -1, -1);
    }
    __syncthreads();

    for (int task = tid; task < BROWS * 40; task += 256) {  // 40 chunks of 8 cover K=0..319
        int r = task / 40, c = task % 40;
        float acc[8] = {0.f, 0.f, 0.f, 0.f, 0.f, 0.f, 0.f, 0.f};
        if (c < 38) {                                       // cols 0..303 sourced from h rows
            int4 mp = Map[r];
            int idx[4] = {mp.x, mp.y, mp.z, mp.w};
            #pragma unroll
            for (int j = 0; j < 4; j++) {
                if (idx[j] >= 0) {
                    uint4 q = *(const uint4*)(h_in + (size_t)idx[j] * H_STRIDE + c * 8);
                    bf2_acc(q.x, acc[0], acc[1]);
                    bf2_acc(q.y, acc[2], acc[3]);
                    bf2_acc(q.z, acc[4], acc[5]);
                    bf2_acc(q.w, acc[6], acc[7]);
                }
            }
        }
        alignas(16) unsigned int o[4];
        #pragma unroll
        for (int e = 0; e < 4; e++) o[e] = f2bf_pk(acc[2 * e], acc[2 * e + 1]);
        *(uint4*)&Alds[r * K2_STRIDE + c * 8] = *(const uint4*)o;
    }
    __syncthreads();

    const int wave = tid >> 5, lane = tid & 31;
    for (int nt = wave; nt < NT_TILES; nt += 8) {
        v16bf b[10];
        #pragma unroll
        for (int k = 0; k < 10; k++) b[k] = load_b_frag(WbH, K2_STRIDE, nt, k * 32, lane);
        for (int mt = 0; mt < BROWS / 16; mt++) {
            v8f acc = {};
            #pragma unroll
            for (int k = 0; k < 10; k++) {
                v16bf a = load_a_frag(Alds, K2_STRIDE, mt * 16, k * 32, lane);
                acc = __builtin_amdgcn_wmma_f32_16x16x32_bf16(false, a, false, b[k],
                                                              (short)0, acc, false, false);
            }
            int col = nt * 16 + (lane & 15);
            int hi  = lane >> 4;
            int row0 = blk * BROWS + mt * 16 + hi * 8;
            size_t base = (size_t)row0 * H_STRIDE + col;
            if (full) {
                #pragma unroll
                for (int rr = 0; rr < 8; rr++) {
                    size_t off = base + (size_t)rr * H_STRIDE;
                    float v = bf2f(inp[off]) + acc[rr];     // pad cols: inp=0, acc=0 -> stays 0
                    h_out[off] = f2bf(fmaxf(v, 0.f));
                }
            } else {
                #pragma unroll
                for (int rr = 0; rr < 8; rr++) {
                    if (row0 + rr < NB1) {
                        size_t off = base + (size_t)rr * H_STRIDE;
                        float v = bf2f(inp[off]) + acc[rr];
                        h_out[off] = f2bf(fmaxf(v, 0.f));
                    }
                }
            }
        }
    }
}

// ---------- kernel 3: atom readout + per-molecule mean ----------
__global__ __launch_bounds__(256) void k3_readout(const int* a2b, const unsigned short* h,
                                                  const float* atom_feat, const unsigned short* WbO,
                                                  const float* b_o, float* out) {
    __shared__ __align__(16) unsigned short Alds[64 * K3_STRIDE];
    __shared__ int4 Map[64];
    const int tid = threadIdx.x, blk = blockIdx.x;

    if (tid < 64) Map[tid] = ((const int4*)a2b)[blk * 64 + tid];   // atoms: no partial blocks
    __syncthreads();

    for (int task = tid; task < 64 * 56; task += 256) {     // 56 chunks of 8 cover K=0..447
        int r = task / 56, c = task % 56;
        int g = blk * 64 + r;
        int4 mp = Map[r];
        int idx[4] = {mp.x, mp.y, mp.z, mp.w};
        alignas(16) unsigned int o[4];
        if (c < 37) {                                       // pure gather: cols 0..295 (messages)
            float acc[8] = {0.f, 0.f, 0.f, 0.f, 0.f, 0.f, 0.f, 0.f};
            #pragma unroll
            for (int j = 0; j < 4; j++) {
                if (idx[j] >= 0) {
                    uint4 q = *(const uint4*)(h + (size_t)idx[j] * H_STRIDE + c * 8);
                    bf2_acc(q.x, acc[0], acc[1]);
                    bf2_acc(q.y, acc[2], acc[3]);
                    bf2_acc(q.z, acc[4], acc[5]);
                    bf2_acc(q.w, acc[6], acc[7]);
                }
            }
            #pragma unroll
            for (int e = 0; e < 4; e++) o[e] = f2bf_pk(acc[2 * e], acc[2 * e + 1]);
        } else {                                            // mixed tail: cols 296..447
            float t[8];
            #pragma unroll
            for (int k = 0; k < 8; k++) {
                int col = c * 8 + k;
                float v = 0.f;
                if (col < D_HIDDEN) {                       // cols 296..299: scalar gather
                    #pragma unroll
                    for (int j = 0; j < 4; j++)
                        if (idx[j] >= 0) v += bf2f(h[(size_t)idx[j] * H_STRIDE + col]);
                } else if (col < 433) {                     // atom features
                    v = atom_feat[(size_t)g * ATOM_FDIM + (col - 300)];
                }
                t[k] = v;
            }
            #pragma unroll
            for (int e = 0; e < 4; e++) o[e] = f2bf_pk(t[2 * e], t[2 * e + 1]);
        }
        *(uint4*)&Alds[r * K3_STRIDE + c * 8] = *(const uint4*)o;
    }
    __syncthreads();

    const int wave = tid >> 5, lane = tid & 31;
    for (int nt = wave; nt < NT_TILES; nt += 8) {
        v16bf b[14];
        #pragma unroll
        for (int k = 0; k < 14; k++) b[k] = load_b_frag(WbO, K3_STRIDE, nt, k * 32, lane);
        for (int mt = 0; mt < 4; mt++) {                    // one m-tile == one molecule (16 atoms)
            v8f acc = {};
            #pragma unroll
            for (int k = 0; k < 14; k++) {
                v16bf a = load_a_frag(Alds, K3_STRIDE, mt * 16, k * 32, lane);
                acc = __builtin_amdgcn_wmma_f32_16x16x32_bf16(false, a, false, b[k],
                                                              (short)0, acc, false, false);
            }
            int col = nt * 16 + (lane & 15);
            float bo = (col < D_HIDDEN) ? b_o[col] : 0.f;   // guard: b_o has 300 entries
            float s = 0.f;
            #pragma unroll
            for (int rr = 0; rr < 8; rr++)
                s += fmaxf(acc[rr] + bo, 0.f);              // cols>=300: acc=0, bo=0 -> relu(0)=0
            s += __shfl_xor(s, 16, 32);                     // + other 8 rows (lanes L <-> L+16)
            if (lane < 16 && col < D_HIDDEN) {
                int mol = blk * 4 + mt;
                out[(size_t)mol * OUT_DIM + col] = s * (1.f / 16.f);
            }
        }
    }
}

// ---------- kernel 4: concat global features ----------
__global__ __launch_bounds__(256) void k4_copy_gf(const float* gf, float* out) {
    size_t i = (size_t)blockIdx.x * 256 + threadIdx.x;
    if (i < (size_t)N_MOLS * GF_DIM) {
        size_t mol = i / GF_DIM, j = i % GF_DIM;
        out[mol * OUT_DIM + D_HIDDEN + j] = gf[i];
    }
}

extern "C" void kernel_launch(void* const* d_in, const int* in_sizes, int n_in,
                              void* d_out, int out_size, void* d_ws, size_t ws_size,
                              hipStream_t stream) {
    const float* atom_feat = (const float*)d_in[0];
    const float* f_ini     = (const float*)d_in[1];
    const int*   a2b       = (const int*)d_in[2];
    const int*   mapping   = (const int*)d_in[3];
    const float* gf        = (const float*)d_in[4];
    const float* W_i       = (const float*)d_in[5];
    const float* W_h       = (const float*)d_in[6];
    const float* W_o       = (const float*)d_in[7];
    const float* b_o       = (const float*)d_in[8];
    float* out = (float*)d_out;

    char* ws = (char*)d_ws;
    const size_t hbytes = (size_t)NB1 * H_STRIDE * sizeof(unsigned short);   // 16B-aligned
    unsigned short* inp = (unsigned short*)ws;  ws += hbytes;
    unsigned short* hA  = (unsigned short*)ws;  ws += hbytes;
    unsigned short* hB  = (unsigned short*)ws;  ws += hbytes;
    unsigned short* WbI = (unsigned short*)ws;  ws += (size_t)304 * K1_STRIDE * 2;
    unsigned short* WbH = (unsigned short*)ws;  ws += (size_t)304 * K2_STRIDE * 2;
    unsigned short* WbO = (unsigned short*)ws;  ws += (size_t)304 * K3_STRIDE * 2;

    prep_weights<<<(304 * K3_STRIDE + 255) / 256, 256, 0, stream>>>(W_i, W_h, W_o, WbI, WbH, WbO);

    const int bond_blocks = (NB1 + BROWS - 1) / BROWS;   // 1366
    k1_init<<<bond_blocks, 256, 0, stream>>>(f_ini, WbI, inp, hA);
    k2_msg<<<bond_blocks, 256, 0, stream>>>(mapping, hA, inp, WbH, hB);   // depth iter 1
    k2_msg<<<bond_blocks, 256, 0, stream>>>(mapping, hB, inp, WbH, hA);   // depth iter 2
    k3_readout<<<NUM_ATOMS / 64, 256, 0, stream>>>(a2b, hA, atom_feat, WbO, b_o, out);
    k4_copy_gf<<<(int)(((size_t)N_MOLS * GF_DIM + 255) / 256), 256, 0, stream>>>(gf, out);
}
